// MaskedMultiSelfAttention_3917010174650
// MI455X (gfx1250) — compile-verified, matrix operand order checked by reading, not run on hardware
//
#include <hip/hip_runtime.h>
#include <hip/hip_bf16.h>

// Problem constants (B,T,C,N from reference)
#define BB 2
#define TT 2048
#define CC 1024
#define NH 16
#define DD 64
#define C3 (3*CC)

typedef __attribute__((ext_vector_type(16))) _Float16 v16h;
typedef __attribute__((ext_vector_type(8)))  _Float16 v8h;
typedef __attribute__((ext_vector_type(8)))  float    v8f;

// Load a 16x32 f16 operand tile in the CDNA5 WMMA A-operand layout.
// Lanes 0-15: row M=lane, halves [k0..k0+7] then [k0+16..k0+23]
// Lanes 16-31: row M=lane-16, halves [k0+8..k0+15] then [k0+24..k0+31]
// Used for both A operands (row-major source) and B operands
// (source pre-transposed so "row" == output column).
__device__ __forceinline__ v16h load_frag(const _Float16* __restrict__ base,
                                          int row0, int k0, int ld) {
    int lane = threadIdx.x & 31;
    int r  = lane & 15;
    int kk = k0 + ((lane >> 4) << 3);
    const _Float16* p = base + (size_t)(row0 + r) * (size_t)ld + kk;
    v8h lo = *(const v8h*)(p);
    v8h hi = *(const v8h*)(p + 16);
    v16h out;
#pragma unroll
    for (int i = 0; i < 8; ++i) { out[i] = lo[i]; out[i + 8] = hi[i]; }
    return out;
}

// ---------------- conversion kernels ----------------
__global__ void cvt_f32_to_f16(const float* __restrict__ src,
                               _Float16* __restrict__ dst, int n) {
    int i = blockIdx.x * 256 + threadIdx.x;
    if (i < n) dst[i] = (_Float16)src[i];
}

// src: rows x cols row-major (fp32) -> dst: cols x rows row-major (f16)
__global__ void cvt_f32_to_f16_T(const float* __restrict__ src,
                                 _Float16* __restrict__ dst, int rows, int cols) {
    int i = blockIdx.x * 256 + threadIdx.x;
    if (i < rows * cols) {
        int r = i / cols, c = i % cols;
        dst[(size_t)c * rows + r] = (_Float16)src[i];
    }
}

// ---------------- QKV GEMM:  (B*T x C) @ (C x 3C) + bias ----------------
// One wave per 64x64 macro-tile (4x4 register blocking of 16x16 WMMA tiles).
// Writes Q,K row-major (b,h,t,d); V transposed (b,h,d,t).
__global__ void __launch_bounds__(256) qkv_gemm_wmma(
        const _Float16* __restrict__ xh, const _Float16* __restrict__ wqkvT,
        const float* __restrict__ bqkv,
        _Float16* __restrict__ Qb, _Float16* __restrict__ Kb,
        _Float16* __restrict__ Vt) {
    const int NT = C3 / 64;                       // 48 col macro-tiles
    int wave = blockIdx.x * 8 + (threadIdx.x >> 5);
    int mt = wave / NT, nt = wave % NT;
    int row0 = mt * 64, col0 = nt * 64;
    int lane = threadIdx.x & 31;

    v8f acc[4][4] = {};
    for (int k0 = 0; k0 < CC; k0 += 32) {
        __builtin_prefetch(xh + (size_t)row0 * CC + k0 + 64, 0, 1);
        v16h a[4], b[4];
#pragma unroll
        for (int i = 0; i < 4; ++i) a[i] = load_frag(xh,    row0 + 16 * i, k0, CC);
#pragma unroll
        for (int j = 0; j < 4; ++j) b[j] = load_frag(wqkvT, col0 + 16 * j, k0, CC);
#pragma unroll
        for (int i = 0; i < 4; ++i)
#pragma unroll
            for (int j = 0; j < 4; ++j)
                acc[i][j] = __builtin_amdgcn_wmma_f32_16x16x32_f16(
                                false, a[i], false, b[j], (short)0, acc[i][j], false, false);
    }

    int cn = lane & 15, hi = lane >> 4;
#pragma unroll
    for (int j = 0; j < 4; ++j) {
        int col   = col0 + 16 * j + cn;
        float bias = bqkv[col];
        int which = col / CC;            // 0=q 1=k 2=v
        int cc    = col % CC;
        int head  = cc / DD, d = cc % DD;
#pragma unroll
        for (int i = 0; i < 4; ++i) {
#pragma unroll
            for (int r = 0; r < 8; ++r) {
                int grow = row0 + 16 * i + r + 8 * hi;   // token index in [0, B*T)
                int bb = grow / TT, t = grow % TT;
                _Float16 h = (_Float16)(acc[i][j][r] + bias);
                size_t hb = (size_t)bb * NH + head;
                if (which == 0)      Qb[(hb * TT + t) * DD + d] = h;
                else if (which == 1) Kb[(hb * TT + t) * DD + d] = h;
                else                 Vt[(hb * DD + d) * TT + t] = h;
            }
        }
    }
}

// ---------------- streaming attention (dense softmax, faithful 1e-9 term) ----------------
// One wave per (b, head, 32-row query block): two 16-row q-tiles share all K/V fragments.
// Online softmax over 32-key chunks.
__global__ void __launch_bounds__(256) attn_wmma(
        const _Float16* __restrict__ Qb, const _Float16* __restrict__ Kb,
        const _Float16* __restrict__ Vt, _Float16* __restrict__ Ab) {
    __shared__ __align__(16) _Float16 plds[8][2 * 16 * 32];   // per-wave P tiles (2 q-tiles)
    int wid  = threadIdx.x >> 5;
    int lane = threadIdx.x & 31;
    int wave = blockIdx.x * 8 + wid;

    int qblk = wave & 63;                  // T/32 = 64 query blocks
    int bh   = wave >> 6;                  // 0..31
    int head = bh & (NH - 1);
    int bb   = bh >> 4;

    size_t hoff = ((size_t)bb * NH + head);
    const _Float16* Qh  = Qb + hoff * TT * DD;
    const _Float16* Kh  = Kb + hoff * TT * DD;
    const _Float16* Vth = Vt + hoff * DD * TT;
    int t0 = qblk * 32;

    v16h qa[2][2];
#pragma unroll
    for (int q = 0; q < 2; ++q) {
        qa[q][0] = load_frag(Qh, t0 + 16 * q, 0,  DD);  // K = d 0..31
        qa[q][1] = load_frag(Qh, t0 + 16 * q, 32, DD);  // K = d 32..63
    }

    v8f o[2][4] = {};
    float m[2][8], lsum[2][8];
#pragma unroll
    for (int q = 0; q < 2; ++q)
#pragma unroll
        for (int r = 0; r < 8; ++r) { m[q][r] = -1e30f; lsum[q][r] = 0.0f; }

    int cn = lane & 15, hi = lane >> 4;
    const float scale = 0.125f;            // 1/sqrt(D)

    for (int s0 = 0; s0 < TT; s0 += 32) {
        // ---- shared K fragments for this 32-key chunk ----
        v16h kb[2][2];
#pragma unroll
        for (int j = 0; j < 2; ++j) {
            kb[j][0] = load_frag(Kh, s0 + 16 * j, 0,  DD);
            kb[j][1] = load_frag(Kh, s0 + 16 * j, 32, DD);
        }
        // ---- shared V fragments (V^T rows are output columns d) ----
        v16h vb[4];
#pragma unroll
        for (int k = 0; k < 4; ++k) vb[k] = load_frag(Vth, 16 * k, s0, TT);

#pragma unroll
        for (int q = 0; q < 2; ++q) {
            int tq0 = t0 + 16 * q;
            // ---- S = Q K^T (two 16x16 D tiles) ----
            v8f s[2];
#pragma unroll
            for (int j = 0; j < 2; ++j) {
                v8f sa = {};
                sa = __builtin_amdgcn_wmma_f32_16x16x32_f16(
                         false, qa[q][0], false, kb[j][0], (short)0, sa, false, false);
                sa = __builtin_amdgcn_wmma_f32_16x16x32_f16(
                         false, qa[q][1], false, kb[j][1], (short)0, sa, false, false);
                s[j] = sa;
            }
            // ---- scale + faithful (1-mask)*1e-9 term + row max ----
            float mrow[8];
#pragma unroll
            for (int r = 0; r < 8; ++r) {
                int trow = tq0 + r + 8 * hi;
#pragma unroll
                for (int j = 0; j < 2; ++j) {
                    int scol = s0 + 16 * j + cn;
                    float v = s[j][r] * scale;
                    if (scol > trow) v += 1e-9f;   // reference adds epsilon, NOT -inf
                    s[j][r] = v;
                }
                mrow[r] = fmaxf(s[0][r], s[1][r]);
            }
#pragma unroll
            for (int msk = 1; msk <= 8; msk <<= 1)
#pragma unroll
                for (int r = 0; r < 8; ++r)
                    mrow[r] = fmaxf(mrow[r], __shfl_xor(mrow[r], msk, 32));

            float alpha[8];
#pragma unroll
            for (int r = 0; r < 8; ++r) {
                float mn = fmaxf(m[q][r], mrow[r]);
                alpha[r] = __expf(m[q][r] - mn);
                m[q][r] = mn;
            }
            // ---- P = exp(S - m), row sums ----
            float rsum[8];
#pragma unroll
            for (int r = 0; r < 8; ++r) {
                float p0 = __expf(s[0][r] - m[q][r]);
                float p1 = __expf(s[1][r] - m[q][r]);
                s[0][r] = p0; s[1][r] = p1;
                rsum[r] = p0 + p1;
            }
#pragma unroll
            for (int msk = 1; msk <= 8; msk <<= 1)
#pragma unroll
                for (int r = 0; r < 8; ++r)
                    rsum[r] += __shfl_xor(rsum[r], msk, 32);
#pragma unroll
            for (int r = 0; r < 8; ++r) lsum[q][r] = lsum[q][r] * alpha[r] + rsum[r];
            // ---- rescale running O ----
#pragma unroll
            for (int k = 0; k < 4; ++k)
#pragma unroll
                for (int r = 0; r < 8; ++r)
                    o[q][k][r] *= alpha[r];
            // ---- relayout P (D-layout -> A-layout) through per-wave LDS ----
            _Float16* pt = &plds[wid][q * 512];
#pragma unroll
            for (int r = 0; r < 8; ++r) {
                int prow = r + 8 * hi;
                pt[prow * 32 + cn]      = (_Float16)s[0][r];
                pt[prow * 32 + 16 + cn] = (_Float16)s[1][r];
            }
        }
        // ---- O += P @ V for both q-tiles, reusing vb ----
#pragma unroll
        for (int q = 0; q < 2; ++q) {
            v16h pa = load_frag(&plds[wid][q * 512], 0, 0, 32);
#pragma unroll
            for (int k = 0; k < 4; ++k)
                o[q][k] = __builtin_amdgcn_wmma_f32_16x16x32_f16(
                              false, pa, false, vb[k], (short)0, o[q][k], false, false);
        }
    }

    // ---- normalize and write attn output row-major (b, t, head*D + d) as f16 ----
#pragma unroll
    for (int q = 0; q < 2; ++q) {
#pragma unroll
        for (int r = 0; r < 8; ++r) {
            float inv = 1.0f / lsum[q][r];
            int t = t0 + 16 * q + r + 8 * hi;
            size_t rowoff = ((size_t)bb * TT + t) * CC + head * DD;
#pragma unroll
            for (int k = 0; k < 4; ++k)
                Ab[rowoff + 16 * k + cn] = (_Float16)(o[q][k][r] * inv);
        }
    }
}

// ---------------- projection GEMM: (B*T x C) @ (C x C) + bias -> fp32 out ----------------
// One wave per 64x64 macro-tile (4x4 register blocking).
__global__ void __launch_bounds__(256) proj_gemm_wmma(
        const _Float16* __restrict__ Ab, const _Float16* __restrict__ wprojT,
        const float* __restrict__ bproj, float* __restrict__ out) {
    const int NT = CC / 64;                       // 16 col macro-tiles
    int wave = blockIdx.x * 8 + (threadIdx.x >> 5);
    int mt = wave / NT, nt = wave % NT;
    int row0 = mt * 64, col0 = nt * 64;
    int lane = threadIdx.x & 31;

    v8f acc[4][4] = {};
    for (int k0 = 0; k0 < CC; k0 += 32) {
        __builtin_prefetch(Ab + (size_t)row0 * CC + k0 + 64, 0, 1);
        v16h a[4], b[4];
#pragma unroll
        for (int i = 0; i < 4; ++i) a[i] = load_frag(Ab,     row0 + 16 * i, k0, CC);
#pragma unroll
        for (int j = 0; j < 4; ++j) b[j] = load_frag(wprojT, col0 + 16 * j, k0, CC);
#pragma unroll
        for (int i = 0; i < 4; ++i)
#pragma unroll
            for (int j = 0; j < 4; ++j)
                acc[i][j] = __builtin_amdgcn_wmma_f32_16x16x32_f16(
                                false, a[i], false, b[j], (short)0, acc[i][j], false, false);
    }
    int cn = lane & 15, hi = lane >> 4;
#pragma unroll
    for (int j = 0; j < 4; ++j) {
        float bias = bproj[col0 + 16 * j + cn];
#pragma unroll
        for (int i = 0; i < 4; ++i) {
#pragma unroll
            for (int r = 0; r < 8; ++r) {
                int grow = row0 + 16 * i + r + 8 * hi;
                out[(size_t)grow * CC + col0 + 16 * j + cn] = acc[i][j][r] + bias;
            }
        }
    }
}

extern "C" void kernel_launch(void* const* d_in, const int* in_sizes, int n_in,
                              void* d_out, int out_size, void* d_ws, size_t ws_size,
                              hipStream_t stream) {
    (void)in_sizes; (void)n_in; (void)out_size; (void)ws_size;
    const float* x     = (const float*)d_in[0];
    const float* Wqkv  = (const float*)d_in[1];
    const float* bqkv  = (const float*)d_in[2];
    const float* Wproj = (const float*)d_in[3];
    const float* bproj = (const float*)d_in[4];
    float* out = (float*)d_out;

    // workspace carve-up (f16 elements)
    _Float16* ws = (_Float16*)d_ws;
    size_t o = 0;
    _Float16* Xh     = ws + o; o += (size_t)BB * TT * CC;       // 4M
    _Float16* WqkvT  = ws + o; o += (size_t)C3 * CC;            // 3M
    _Float16* WprojT = ws + o; o += (size_t)CC * CC;            // 1M
    _Float16* Qb     = ws + o; o += (size_t)BB * NH * TT * DD;  // 4M
    _Float16* Kb     = ws + o; o += (size_t)BB * NH * TT * DD;  // 4M
    _Float16* Vt     = ws + o; o += (size_t)BB * NH * DD * TT;  // 4M
    _Float16* Ab     = ws + o;                                  // 4M

    int nx = BB * TT * CC;
    cvt_f32_to_f16<<<(nx + 255) / 256, 256, 0, stream>>>(x, Xh, nx);
    int nw1 = CC * C3;
    cvt_f32_to_f16_T<<<(nw1 + 255) / 256, 256, 0, stream>>>(Wqkv, WqkvT, CC, C3);
    int nw2 = CC * CC;
    cvt_f32_to_f16_T<<<(nw2 + 255) / 256, 256, 0, stream>>>(Wproj, WprojT, CC, CC);

    int waves_qkv  = (BB * TT / 64) * (C3 / 64);   // 64*48 = 3072 waves
    qkv_gemm_wmma<<<waves_qkv / 8, 256, 0, stream>>>(Xh, WqkvT, bqkv, Qb, Kb, Vt);

    int waves_attn = BB * NH * (TT / 32);          // 2048 waves
    attn_wmma<<<waves_attn / 8, 256, 0, stream>>>(Qb, Kb, Vt, Ab);

    int waves_proj = (BB * TT / 64) * (CC / 64);   // 64*16 = 1024 waves
    proj_gemm_wmma<<<waves_proj / 8, 256, 0, stream>>>(Ab, WprojT, bproj, out);
}
